// SelfAttention_10977936408948
// MI455X (gfx1250) — matrix operand order repaired and checked by reading, not compiled, not run on hardware
//
#include <hip/hip_runtime.h>

typedef __attribute__((ext_vector_type(16))) _Float16 v16h;
typedef __attribute__((ext_vector_type(8)))  _Float16 v8h;
typedef __attribute__((ext_vector_type(8)))  float    v8f;

union Frag { v16h v; v8h h[2]; };

#define BATCH  4
#define SEQ    2048
#define DMODEL 1024
#define DK     64
#define NSLAB  (DMODEL / 32)   // 32 K-slabs for the projection
#define RTILES (SEQ / 16)      // 128 row tiles per batch

// LDS row pitches (in halves) chosen so 16-lane strided reads hit 16
// distinct banks: K rows 72 (36 dwords), V/P rows 40 (20 dwords).
#define KPITCH 72
#define VPITCH 40
#define PPITCH 40

__device__ __forceinline__ void sched_fence() {
#if __has_builtin(__builtin_amdgcn_sched_barrier)
  __builtin_amdgcn_sched_barrier(0);
#endif
}

// ---------------------------------------------------------------------------
// Kernel 1: pack Wq/Wk/Wv into WMMA B-fragment layout (f16).
// wp[((m*32+slab)*4+nt)*512 + lane*16 + t] =
//   W_m[(slab*32 + (t&8 ? 16 : 0) + (lane<16 ? 0 : 8) + (t&7)) * 64 + nt*16 + (lane&15)]
// ---------------------------------------------------------------------------
__global__ void pack_w(const float* __restrict__ Wq, const float* __restrict__ Wk,
                       const float* __restrict__ Wv, _Float16* __restrict__ wp) {
  int idx  = blockIdx.x * blockDim.x + threadIdx.x;   // 196608 threads
  int t    = idx & 15;
  int lane = (idx >> 4) & 31;
  int nt   = (idx >> 9) & 3;
  int slab = (idx >> 11) & 31;
  int m    = idx >> 16;
  const float* W = (m == 0) ? Wq : (m == 1) ? Wk : Wv;
  int k   = slab * 32 + ((t & 8) ? 16 : 0) + ((lane < 16) ? 0 : 8) + (t & 7);
  int col = nt * 16 + (lane & 15);
  wp[idx] = (_Float16)W[k * DK + col];
}

// ---------------------------------------------------------------------------
// Kernel 2: QKV projection, fp32->f16 conversion fused into the A-fragment
// load (x is read exactly once from HBM). One wave per 16-row tile.
// A sched-barrier pins all 12 B-fragment loads before the 12 WMMAs so the
// loop issues one load clause + one wait per 12 matrix ops.
// Q,K stored row-major f16 [b, s, 64]; V stored transposed f16 [b, 64, s].
// ---------------------------------------------------------------------------
__global__ void __launch_bounds__(128)
qkv_kernel(const float* __restrict__ x, const _Float16* __restrict__ wp,
           _Float16* __restrict__ Qh, _Float16* __restrict__ Kh,
           _Float16* __restrict__ Vt) {
  int lane = threadIdx.x & 31;
  int wave = threadIdx.x >> 5;
  int tile = blockIdx.x * 4 + wave;        // 0..511
  int b    = tile >> 7;
  int rt   = tile & 127;
  int rowbase = rt * 16;
  int l15 = lane & 15;
  int hi  = (lane >= 16) ? 1 : 0;

  v8f zf = {};
  v8f acc[3][4];
#pragma unroll
  for (int m = 0; m < 3; ++m)
#pragma unroll
    for (int nt = 0; nt < 4; ++nt) acc[m][nt] = zf;

  // A-fragment base (fp32): row = rowbase + l%16; K runs per WMMA A layout.
  const float* arow =
      x + (size_t)(b * SEQ + rowbase + l15) * DMODEL + (hi ? 8 : 0);

  for (int slab = 0; slab < NSLAB; ++slab) {
    // Load fp32, convert to the f16 A fragment in registers.
    v8f f0 = *(const v8f*)(arow + slab * 32);        // K slab+[0..8)/[8..16)
    v8f f1 = *(const v8f*)(arow + slab * 32 + 16);   // K slab+[16..24)/[24..32)
    Frag a;
#pragma unroll
    for (int i = 0; i < 8; ++i) {
      a.h[0][i] = (_Float16)f0[i];
      a.h[1][i] = (_Float16)f1[i];
    }
    // Batch all 12 B fragments for this slab into live registers.
    Frag bf[12];
#pragma unroll
    for (int m = 0; m < 3; ++m)
#pragma unroll
      for (int nt = 0; nt < 4; ++nt)
        bf[m * 4 + nt].v = *(const v16h*)(
            wp + ((size_t)((m * NSLAB + slab) * 4 + nt) * 32 + lane) * 16);
    sched_fence();   // loads above, WMMAs below: one clause + one wait per 12
    // 12 WMMAs back to back.
#pragma unroll
    for (int m = 0; m < 3; ++m)
#pragma unroll
      for (int nt = 0; nt < 4; ++nt)
        acc[m][nt] = __builtin_amdgcn_wmma_f32_16x16x32_f16(
            false, a.v, false, bf[m * 4 + nt].v, (short)0, acc[m][nt],
            false, false);
  }

  // Epilogue: D layout -> memory. VGPR i holds row i (lanes<16) / i+8.
#pragma unroll
  for (int nt = 0; nt < 4; ++nt) {
    int col = nt * 16 + l15;
#pragma unroll
    for (int i = 0; i < 8; ++i) {
      int row = rowbase + i + (hi ? 8 : 0);
      Qh[(size_t)(b * SEQ + row) * DK + col] = (_Float16)acc[0][nt][i];
      Kh[(size_t)(b * SEQ + row) * DK + col] = (_Float16)acc[1][nt][i];
    }
    v8h vv;
#pragma unroll
    for (int i = 0; i < 8; ++i) vv[i] = (_Float16)acc[2][nt][i];
    *(v8h*)(Vt + (size_t)(b * DK + col) * SEQ + rowbase + (hi ? 8 : 0)) = vv;
  }
}

// ---------------------------------------------------------------------------
// Async DMA helpers: global -> LDS (tracked by per-wave ASYNCcnt). The LDS
// byte offset is the low 32 bits of the generic pointer (LDS aperture).
// Each helper moves one 4KB block with the 256 16-byte units spread across
// all 128 threads of the workgroup (2 async instructions per lane).
// ---------------------------------------------------------------------------
__device__ __forceinline__ void issue_k_copy(unsigned ldsbase,
                                             const _Float16* gsrc, int tid) {
  // 32 rows x 64 halves contiguous in global; LDS rows padded to KPITCH.
#pragma unroll
  for (int i = 0; i < 2; ++i) {
    int unit = i * 128 + tid;                       // 16B units, 0..255
    unsigned lds = ldsbase + (unsigned)((unit >> 3) * (KPITCH * 2)
                                        + (unit & 7) * 16);
    const char* g = (const char*)gsrc + (size_t)unit * 16;
    asm volatile("global_load_async_to_lds_b128 %0, %1, off"
                 :: "v"(lds), "v"(g) : "memory");
  }
}

__device__ __forceinline__ void issue_v_copy(unsigned ldsbase,
                                             const _Float16* vtb_j, int tid) {
  // 64 cols x 32 keys; global rows stride SEQ halves; LDS rows VPITCH halves.
#pragma unroll
  for (int i = 0; i < 2; ++i) {
    int unit = i * 128 + tid;
    int col = unit >> 2, seg = unit & 3;
    unsigned lds = ldsbase + (unsigned)(col * (VPITCH * 2) + seg * 16);
    const char* g = (const char*)(vtb_j + (size_t)col * SEQ) + seg * 16;
    asm volatile("global_load_async_to_lds_b128 %0, %1, off"
                 :: "v"(lds), "v"(g) : "memory");
  }
}

// ---------------------------------------------------------------------------
// Kernel 3: flash-style attention. One wave per 16-query tile; all 4 waves
// of a block share one double-buffered K/V staging area in LDS (they stream
// the same key blocks: 4 consecutive q-tiles always lie in one batch).
// Async DMA of block j+32 overlaps with WMMA + online softmax of block j.
// ---------------------------------------------------------------------------
__global__ void __launch_bounds__(128)
attn_kernel(const _Float16* __restrict__ Qh, const _Float16* __restrict__ Kh,
            const _Float16* __restrict__ Vt, float* __restrict__ out) {
  __shared__ __attribute__((aligned(64))) _Float16 Kbuf[2][32 * KPITCH];
  __shared__ __attribute__((aligned(64))) _Float16 Vbuf[2][64 * VPITCH];
  __shared__ __attribute__((aligned(64))) _Float16 Pb[4][16 * PPITCH];

  int tid  = threadIdx.x;
  int lane = tid & 31;
  int wave = tid >> 5;
  int qt   = blockIdx.x * 4 + wave;       // 0..511
  int b    = qt >> 7;                     // same for all 4 waves of a block
  int rt   = qt & 127;
  int qrow = rt * 16;
  int l15  = lane & 15;
  int hi   = (lane >= 16) ? 1 : 0;

  // Q A-fragments (two K=32 slabs covering d_k = 64); stays resident.
  const _Float16* qp = Qh + (size_t)(b * SEQ + qrow + l15) * DK + (hi ? 16 : 0);
  Frag aq0, aq1;
  aq0.h[0] = *(const v8h*)(qp);
  aq0.h[1] = *(const v8h*)(qp + 8);
  aq1.h[0] = *(const v8h*)(qp + 32);
  aq1.h[1] = *(const v8h*)(qp + 40);

  float mrun[8], lrun[8];
#pragma unroll
  for (int i = 0; i < 8; ++i) { mrun[i] = -1e30f; lrun[i] = 0.f; }
  v8f zf = {};
  v8f acc[4];
#pragma unroll
  for (int nt = 0; nt < 4; ++nt) acc[nt] = zf;

  const _Float16* kbase = Kh + (size_t)(b * SEQ) * DK;
  const _Float16* vtb   = Vt + (size_t)(b * DK) * SEQ;
  const float scale = 0.125f;   // 1/sqrt(64)

  unsigned klds[2], vlds[2];
#pragma unroll
  for (int i = 0; i < 2; ++i) {
    klds[i] = (unsigned)(size_t)&Kbuf[i][0];
    vlds[i] = (unsigned)(size_t)&Vbuf[i][0];
  }

  // Prologue: whole block cooperatively DMAs key-block 0 into buffer 0.
  issue_k_copy(klds[0], kbase, tid);
  issue_v_copy(vlds[0], vtb, tid);

  for (int j = 0; j < SEQ; j += 32) {
    int cur = (j >> 5) & 1;
    // Each wave drains its own ASYNCcnt, then the barrier publishes buffers.
    asm volatile("s_wait_asynccnt 0x0" ::: "memory");
    __syncthreads();
    // Kick off next block's DMA into the other buffer (safe: buf cur^1 was
    // last read in iteration j-32, completed before the barrier above).
    if (j + 32 < SEQ) {
      issue_k_copy(klds[cur ^ 1], kbase + (size_t)(j + 32) * DK, tid);
      issue_v_copy(vlds[cur ^ 1], vtb + (j + 32), tid);
    }

    // K^T B-fragments from LDS (padded pitch, bank-conflict-free).
    const _Float16* kb = &Kbuf[cur][0];
    const _Float16* k0 = kb + l15 * KPITCH + (hi ? 16 : 0);
    const _Float16* k1 = kb + (16 + l15) * KPITCH + (hi ? 16 : 0);
    Frag bk00, bk01, bk10, bk11;
    bk00.h[0] = *(const v8h*)(k0);       bk00.h[1] = *(const v8h*)(k0 + 8);
    bk01.h[0] = *(const v8h*)(k0 + 32);  bk01.h[1] = *(const v8h*)(k0 + 40);
    bk10.h[0] = *(const v8h*)(k1);       bk10.h[1] = *(const v8h*)(k1 + 8);
    bk11.h[0] = *(const v8h*)(k1 + 32);  bk11.h[1] = *(const v8h*)(k1 + 40);

    // V B-fragments for this key block (independent of softmax -> load early).
    const _Float16* vb = &Vbuf[cur][0];
    Frag bv[4];
#pragma unroll
    for (int nt = 0; nt < 4; ++nt) {
      const _Float16* p = vb + (nt * 16 + l15) * VPITCH + (hi ? 16 : 0);
      bv[nt].h[0] = *(const v8h*)(p);
      bv[nt].h[1] = *(const v8h*)(p + 8);
    }

    v8f s0 = zf, s1 = zf;
    s0 = __builtin_amdgcn_wmma_f32_16x16x32_f16(false, aq0.v, false, bk00.v,
                                                (short)0, s0, false, false);
    s0 = __builtin_amdgcn_wmma_f32_16x16x32_f16(false, aq1.v, false, bk01.v,
                                                (short)0, s0, false, false);
    s1 = __builtin_amdgcn_wmma_f32_16x16x32_f16(false, aq0.v, false, bk10.v,
                                                (short)0, s1, false, false);
    s1 = __builtin_amdgcn_wmma_f32_16x16x32_f16(false, aq1.v, false, bk11.v,
                                                (short)0, s1, false, false);

    // Online softmax. Row M lives in (VGPR i, 16-lane half); xor-shuffles
    // with masks 8..1 reduce across the 16 lanes of each half (wave32).
#pragma unroll
    for (int i = 0; i < 8; ++i) {
      float a0 = s0[i] * scale;
      float a1 = s1[i] * scale;
      float mx = fmaxf(a0, a1);
#pragma unroll
      for (int off = 8; off > 0; off >>= 1)
        mx = fmaxf(mx, __shfl_xor(mx, off, 32));
      float newm  = fmaxf(mrun[i], mx);
      float alpha = __expf(mrun[i] - newm);
      mrun[i] = newm;
      float p0 = __expf(a0 - newm);
      float p1 = __expf(a1 - newm);
      float rs = p0 + p1;
#pragma unroll
      for (int off = 8; off > 0; off >>= 1)
        rs += __shfl_xor(rs, off, 32);
      lrun[i] = lrun[i] * alpha + rs;
#pragma unroll
      for (int nt = 0; nt < 4; ++nt) acc[nt][i] *= alpha;
      int m = i + (hi ? 8 : 0);
      Pb[wave][m * PPITCH + l15]      = (_Float16)p0;   // key cols 0..15
      Pb[wave][m * PPITCH + 16 + l15] = (_Float16)p1;   // key cols 16..31
    }

    // Reload P as an A fragment (same wave wrote it; LDS in-order per wave).
    Frag ap;
    const _Float16* pr = &Pb[wave][l15 * PPITCH] + (hi ? 16 : 0);
    ap.h[0] = *(const v8h*)(pr);
    ap.h[1] = *(const v8h*)(pr + 8);

    // P (16x32) x V-slab (32x64).
#pragma unroll
    for (int nt = 0; nt < 4; ++nt)
      acc[nt] = __builtin_amdgcn_wmma_f32_16x16x32_f16(
          false, ap.v, false, bv[nt].v, (short)0, acc[nt], false, false);
  }

  // Final normalize + store fp32 output.
#pragma unroll
  for (int nt = 0; nt < 4; ++nt) {
    int col = nt * 16 + l15;
#pragma unroll
    for (int i = 0; i < 8; ++i) {
      int row = qrow + i + (hi ? 8 : 0);
      out[(size_t)(b * SEQ + row) * DK + col] = acc[nt][i] / lrun[i];
    }
  }
}

// ---------------------------------------------------------------------------
// Launch
// ---------------------------------------------------------------------------
extern "C" void kernel_launch(void* const* d_in, const int* in_sizes, int n_in,
                              void* d_out, int out_size, void* d_ws,
                              size_t ws_size, hipStream_t stream) {
  (void)in_sizes; (void)n_in; (void)out_size; (void)ws_size;
  const float* x  = (const float*)d_in[0];
  const float* Wq = (const float*)d_in[1];
  const float* Wk = (const float*)d_in[2];
  const float* Wv = (const float*)d_in[3];

  const size_t WP_BYTES  = (size_t)3 * 32 * 4 * 32 * 16 * 2;  // 384 KB
  const size_t QKV_BYTES = (size_t)BATCH * SEQ * DK * 2;      // 1 MB each

  char* ws = (char*)d_ws;
  _Float16* wp = (_Float16*)(ws);
  _Float16* Qh = (_Float16*)(ws + WP_BYTES);
  _Float16* Kh = (_Float16*)(ws + WP_BYTES + QKV_BYTES);
  _Float16* Vt = (_Float16*)(ws + WP_BYTES + 2 * QKV_BYTES);

  pack_w<<<196608 / 256, 256, 0, stream>>>(Wq, Wk, Wv, wp);
  qkv_kernel<<<(BATCH * RTILES) / 4, 128, 0, stream>>>(x, wp, Qh, Kh, Vt);
  attn_kernel<<<(BATCH * RTILES) / 4, 128, 0, stream>>>(Qh, Kh, Vt,
                                                        (float*)d_out);
}
